// MultiHeadAttention_34866544509380
// MI455X (gfx1250) — compile-verified, matrix operand order checked by reading, not run on hardware
//
#include <hip/hip_runtime.h>
#include <hip/hip_bf16.h>

// ---------------------------------------------------------------------------
// MHA forward for MI455X (gfx1250): wave32, WMMA f16 path, TDM-staged V panels.
// d_out = [attn_output (2*2048*1024 f32)] ++ [attn_weights (2*16*2048*2048 f32)]
// ---------------------------------------------------------------------------

typedef __attribute__((ext_vector_type(16))) _Float16 v16h;
typedef __attribute__((ext_vector_type(8)))  float    v8f;
typedef unsigned int u32x4 __attribute__((ext_vector_type(4)));
typedef int          i32x8 __attribute__((ext_vector_type(8)));
typedef int          i32x4 __attribute__((ext_vector_type(4)));

#define E_DIM   1024
#define H_NUM   16
#define DH      64
#define S_LEN   2048
#define B_NUM   2
#define VCHUNK  256          // V rows staged per TDM descriptor (32 KB f16)

// ---- WMMA fragment helpers (ISA 7.12.2 layouts, wave32) -------------------
__device__ __forceinline__ v16h load_frag(const _Float16* tile, int own_stride, int k_stride) {
  const int lane = threadIdx.x & 31;
  const int own  = lane & 15;
  const int h8   = (lane >> 4) << 3;
  const _Float16* p = tile + (size_t)own * own_stride;
  v16h f;
#pragma unroll
  for (int i = 0; i < 8; ++i) f[i]     = p[(size_t)(h8 + i)      * k_stride];
#pragma unroll
  for (int i = 0; i < 8; ++i) f[8 + i] = p[(size_t)(16 + h8 + i) * k_stride];
  return f;
}

__device__ __forceinline__ v8f wmma16(const v16h& a, const v16h& b, const v8f& c) {
  return __builtin_amdgcn_wmma_f32_16x16x32_f16(false, a, false, b, (short)0, c,
                                                false, false);
}

// ---- TDM: DMA a VCHUNKx64 f16 tile (row stride 1024 elems) into LDS -------
// D# built per ISA §8.3/§8.4: group0 = {count, lds_addr, global_addr, type=2},
// group1 = {data_size=2B, tensor_dim0=64, tensor_dim1=2048, tile=64xVCHUNK,
//           dim0_stride=1024}. 2D tensor -> remaining groups zero (6-arg form).
__device__ __forceinline__ void tdm_load_vchunk(const _Float16* gsrc, unsigned lds_byte_off) {
  const unsigned long long ga = (unsigned long long)(size_t)gsrc;
  u32x4 g0;
  g0[0] = 1u;                                            // count=1 (user mode)
  g0[1] = lds_byte_off;                                  // lds_addr (bytes)
  g0[2] = (unsigned)ga;                                  // global_addr[31:0]
  g0[3] = (unsigned)((ga >> 32) & 0x01FFFFFFu) | (2u << 30); // addr[56:32] | type=2
  i32x8 g1;
  g1[0] = (int)(1u << 16);                               // data_size=1 (2 bytes)
  g1[1] = (int)(64u << 16);                              // tensor_dim0 = 64
  g1[2] = (int)(2048u << 16);                            // tensor_dim1 = 2048
  g1[3] = (int)(64u << 16);                              // tile_dim0 = 64
  g1[4] = (int)VCHUNK;                                   // tile_dim1 = 256
  g1[5] = (int)1024;                                     // tensor_dim0_stride = 1024
  g1[6] = 0;
  g1[7] = 0;
  i32x4 z4 = {0, 0, 0, 0};
  i32x8 z8 = {0, 0, 0, 0, 0, 0, 0, 0};
  __builtin_amdgcn_tensor_load_to_lds(g0, g1, z4, z4, z8, 0);
}

// ---- f32 -> f16 conversion -------------------------------------------------
__global__ __launch_bounds__(256) void cvt_f32_to_f16(const float* __restrict__ in,
                                                      _Float16* __restrict__ out, int n) {
  int i = (blockIdx.x * 256 + threadIdx.x) * 4;
  if (i + 3 < n) {
    float4 v = *(const float4*)(in + i);
    out[i + 0] = (_Float16)v.x;
    out[i + 1] = (_Float16)v.y;
    out[i + 2] = (_Float16)v.z;
    out[i + 3] = (_Float16)v.w;
  }
}

// ---- Projection GEMM: out[m,n] = (sum_k X[m,k]*W[n,k] + bias[n]) * scale ---
// Wave computes 16(M)x64(N). Unroll-by-2 ping-pong pipeline: two disjoint
// fragment register sets, each written directly by its loads (no rotation
// copies, no WAR v_nops), loads of one set overlapping WMMAs of the other.
__global__ __launch_bounds__(256) void proj_gemm(const _Float16* __restrict__ X,
                                                 const _Float16* __restrict__ W,
                                                 const float* __restrict__ bias,
                                                 float scale,
                                                 _Float16* __restrict__ outH,
                                                 float* __restrict__ outF,
                                                 int M) {
  const int wave   = blockIdx.x * 8 + (threadIdx.x >> 5);
  const int nStrip = E_DIM / 64;                 // 16
  const int mTile  = wave / nStrip;
  const int n0     = (wave % nStrip) * 64;
  if (mTile * 16 >= M) return;                   // wave-uniform

  const _Float16* Arow = X + (size_t)mTile * 16 * E_DIM;
  const _Float16* Wrow = W + (size_t)n0 * E_DIM;
  v8f acc[4] = {v8f{}, v8f{}, v8f{}, v8f{}};

  v16h a0 = load_frag(Arow, E_DIM, 1);
  v16h b0[4], a1, b1[4];
#pragma unroll
  for (int t = 0; t < 4; ++t)
    b0[t] = load_frag(Wrow + (size_t)t * 16 * E_DIM, E_DIM, 1);

  for (int kk = 0; kk < E_DIM - 64; kk += 64) {
    a1 = load_frag(Arow + kk + 32, E_DIM, 1);
#pragma unroll
    for (int t = 0; t < 4; ++t)
      b1[t] = load_frag(Wrow + (size_t)t * 16 * E_DIM + kk + 32, E_DIM, 1);
#pragma unroll
    for (int t = 0; t < 4; ++t) acc[t] = wmma16(a0, b0[t], acc[t]);
    a0 = load_frag(Arow + kk + 64, E_DIM, 1);
#pragma unroll
    for (int t = 0; t < 4; ++t)
      b0[t] = load_frag(Wrow + (size_t)t * 16 * E_DIM + kk + 64, E_DIM, 1);
#pragma unroll
    for (int t = 0; t < 4; ++t) acc[t] = wmma16(a1, b1[t], acc[t]);
  }
  // tail: steps E-64 and E-32
  a1 = load_frag(Arow + E_DIM - 32, E_DIM, 1);
#pragma unroll
  for (int t = 0; t < 4; ++t)
    b1[t] = load_frag(Wrow + (size_t)t * 16 * E_DIM + E_DIM - 32, E_DIM, 1);
#pragma unroll
  for (int t = 0; t < 4; ++t) acc[t] = wmma16(a0, b0[t], acc[t]);
#pragma unroll
  for (int t = 0; t < 4; ++t) acc[t] = wmma16(a1, b1[t], acc[t]);

  const int lane = threadIdx.x & 31;
  const int nl = lane & 15, mh = (lane >> 4) << 3;
#pragma unroll
  for (int t = 0; t < 4; ++t) {
    const int n = n0 + t * 16 + nl;
    const float bv = bias[n];
#pragma unroll
    for (int r = 0; r < 8; ++r) {
      const float v = (acc[t][r] + bv) * scale;
      const size_t idx = (size_t)(mTile * 16 + mh + r) * E_DIM + n;
      if (outH) outH[idx] = (_Float16)v;
      else      outF[idx] = v;
    }
  }
}

// ---- Attention core: one block = (b, h, 16-row q tile), 8 waves -----------
// LDS: sS 16x2048 f32 (128K) | sP 16x2048 f16 (64K) | sO 16x64 f32 (4K)
//      | sV 2 x (VCHUNK x 64 f16) (64K)  => 260 KB of the WGP's 320 KB
__global__ __launch_bounds__(256) void attn_kernel(const _Float16* __restrict__ Q,
                                                   const _Float16* __restrict__ K,
                                                   const _Float16* __restrict__ V,
                                                   float* __restrict__ attnW,
                                                   _Float16* __restrict__ Ctx) {
  extern __shared__ char smem[];
  float*    sS = (float*)smem;                                   // 16*2048 f32
  _Float16* sP = (_Float16*)(smem + 16 * S_LEN * 4);             // 16*2048 f16
  float*    sO = (float*)(smem + 16 * S_LEN * 4 + 16 * S_LEN * 2);
  _Float16* sV = (_Float16*)(smem + 16 * S_LEN * 4 + 16 * S_LEN * 2 + 16 * 64 * 4);
  const unsigned sV_off = (unsigned)(16 * S_LEN * 4 + 16 * S_LEN * 2 + 16 * 64 * 4);

  const int blk = blockIdx.x;
  const int qt  = blk & (S_LEN / 16 - 1);        // 0..127
  const int h   = (blk >> 7) & (H_NUM - 1);
  const int b   = blk >> 11;
  const int wv  = threadIdx.x >> 5;
  const int lane = threadIdx.x & 31;

  const _Float16* Qbase = Q + ((size_t)(b * S_LEN + qt * 16)) * E_DIM + h * DH;
  const _Float16* Kbase = K + ((size_t)b * S_LEN) * E_DIM + h * DH;
  const _Float16* Vbase = V + ((size_t)b * S_LEN) * E_DIM + h * DH;

  // Kick off the TDM DMA of V chunk 0 immediately: it overlaps phases 1+2.
  if (wv == 0) tdm_load_vchunk(Vbase, sV_off);

  for (int i = threadIdx.x; i < 16 * 64; i += 256) sO[i] = 0.0f;

  // Q fragments for the two 32-wide K-steps of Dh=64 (reused 16x per wave)
  v16h aQ0 = load_frag(Qbase + 0,  E_DIM, 1);
  v16h aQ1 = load_frag(Qbase + 32, E_DIM, 1);

  // ---- Phase 1: S = Q K^T ; wave wv owns columns [wv*256, wv*256+256) ----
  // 2-deep fragment ring (indices constant-fold under full unroll): tile
  // nt+1's K fragments are in flight while tile nt's two WMMAs execute.
  const int nl = lane & 15, mh = (lane >> 4) << 3;
  const _Float16* Kwave = Kbase + (size_t)(wv * 256) * E_DIM;
  v16h kb0[2], kb1[2];
  kb0[0] = load_frag(Kwave + 0,  E_DIM, 1);
  kb1[0] = load_frag(Kwave + 32, E_DIM, 1);
#pragma unroll
  for (int nt = 0; nt < 16; ++nt) {
    if (nt < 15) {
      kb0[(nt + 1) & 1] = load_frag(Kwave + (size_t)(nt + 1) * 16 * E_DIM + 0,  E_DIM, 1);
      kb1[(nt + 1) & 1] = load_frag(Kwave + (size_t)(nt + 1) * 16 * E_DIM + 32, E_DIM, 1);
    }
    v8f acc = v8f{};
    acc = wmma16(aQ0, kb0[nt & 1], acc);
    acc = wmma16(aQ1, kb1[nt & 1], acc);
    const int n0 = wv * 256 + nt * 16;
#pragma unroll
    for (int r = 0; r < 8; ++r) sS[(mh + r) * S_LEN + n0 + nl] = acc[r];
  }
  __syncthreads();

  // ---- Phase 2: softmax per row; wave wv owns rows {2wv, 2wv+1} ----------
  float* attnRow = attnW + (((size_t)(b * H_NUM + h) * S_LEN + qt * 16) * S_LEN);
#pragma unroll
  for (int rr = 0; rr < 2; ++rr) {
    const int row = wv * 2 + rr;
    float* srow = sS + (size_t)row * S_LEN;
    float mx = -3.0e38f;
    for (int c = lane; c < S_LEN; c += 32) mx = fmaxf(mx, srow[c]);
#pragma unroll
    for (int off = 16; off > 0; off >>= 1) mx = fmaxf(mx, __shfl_xor(mx, off, 32));
    float sum = 0.0f;
    for (int c = lane; c < S_LEN; c += 32) {
      float e = __expf(srow[c] - mx);
      srow[c] = e;
      sum += e;
    }
#pragma unroll
    for (int off = 16; off > 0; off >>= 1) sum += __shfl_xor(sum, off, 32);
    const float inv = 1.0f / sum;
    _Float16* prow = sP + (size_t)row * S_LEN;
    float* grow = attnRow + (size_t)row * S_LEN;
    for (int c = lane; c < S_LEN; c += 32) {
      const float p = srow[c] * inv;
      __builtin_nontemporal_store(p, grow + c);  // 537 MB stream: keep out of L2
      prow[c] = (_Float16)p;                     // f16 P for the PV WMMA
    }
  }
  __syncthreads();

  // ---- Phase 3: O = P V over 8 TDM-staged chunks, double-buffered --------
  // wave wv: n-tile = wv>>1, K-half of chunk = wv&1 (4 WMMAs per chunk).
  const int nt  = wv >> 1;
  const int sub = wv & 1;
  v8f oacc = v8f{};
  const int nChunks = S_LEN / VCHUNK;            // 8
  for (int c = 0; c < nChunks; ++c) {
    if (wv == 0) {
      if (c + 1 < nChunks) {
        tdm_load_vchunk(Vbase + (size_t)(c + 1) * VCHUNK * E_DIM,
                        sV_off + (unsigned)(((c + 1) & 1) * VCHUNK * 64 * 2));
        __builtin_amdgcn_s_wait_tensorcnt(1);    // chunk c complete (in-order)
      } else {
        __builtin_amdgcn_s_wait_tensorcnt(0);
      }
    }
    __syncthreads();                             // chunk c visible to all waves
    const _Float16* Vb = sV + (size_t)(c & 1) * VCHUNK * 64;
#pragma unroll
    for (int ks = 0; ks < 4; ++ks) {
      const int kkg = c * VCHUNK + sub * 128 + ks * 32;  // column in P
      const int kkl = sub * 128 + ks * 32;               // row in staged chunk
      v16h a  = load_frag(sP + kkg, S_LEN, 1);
      v16h bf = load_frag(Vb + (size_t)kkl * 64 + nt * 16, 1, 64);
      oacc = wmma16(a, bf, oacc);
    }
    __syncthreads();                             // all done before buf reuse
  }
#pragma unroll
  for (int r = 0; r < 8; ++r)
    atomicAdd(&sO[(mh + r) * 64 + nt * 16 + nl], oacc[r]);
  __syncthreads();

  // ---- Write f16 context [B*S, E] for the output projection --------------
  _Float16* ctx = Ctx + ((size_t)(b * S_LEN + qt * 16)) * E_DIM + h * DH;
  for (int i = threadIdx.x; i < 16 * 64; i += 256) {
    const int r = i >> 6, c = i & 63;
    ctx[(size_t)r * E_DIM + c] = (_Float16)sO[i];
  }
}

// ---------------------------------------------------------------------------
extern "C" void kernel_launch(void* const* d_in, const int* in_sizes, int n_in,
                              void* d_out, int out_size, void* d_ws, size_t ws_size,
                              hipStream_t stream) {
  (void)in_sizes; (void)n_in; (void)out_size; (void)ws_size;
  const float* query = (const float*)d_in[0];
  const float* key   = (const float*)d_in[1];
  const float* value = (const float*)d_in[2];
  const float* Wq = (const float*)d_in[3];  const float* bq = (const float*)d_in[4];
  const float* Wk = (const float*)d_in[5];  const float* bk = (const float*)d_in[6];
  const float* Wv = (const float*)d_in[7];  const float* bv = (const float*)d_in[8];
  const float* Wo = (const float*)d_in[9];  const float* bo = (const float*)d_in[10];

  const int M    = B_NUM * S_LEN;           // 4096 rows
  const int NIN  = M * E_DIM;               // 4,194,304 activation elements
  const int NW   = E_DIM * E_DIM;           // 1,048,576 weight elements

  // Workspace partition (all _Float16), total 64 MB
  _Float16* ws  = (_Float16*)d_ws;
  _Float16* qH  = ws;
  _Float16* kH  = qH  + NIN;
  _Float16* vH  = kH  + NIN;
  _Float16* WqH = vH  + NIN;
  _Float16* WkH = WqH + NW;
  _Float16* WvH = WkH + NW;
  _Float16* WoH = WvH + NW;
  _Float16* Qp  = WoH + NW;
  _Float16* Kp  = Qp  + NIN;
  _Float16* Vp  = Kp  + NIN;
  _Float16* Ctx = Vp  + NIN;

  float* out_attn = (float*)d_out;                         // [B,S,E]
  float* out_w    = (float*)d_out + (size_t)M * E_DIM;     // [B,H,S,S]

  // 1) convert activations + weights to f16
  cvt_f32_to_f16<<<NIN / (256 * 4), 256, 0, stream>>>(query, qH, NIN);
  cvt_f32_to_f16<<<NIN / (256 * 4), 256, 0, stream>>>(key,   kH, NIN);
  cvt_f32_to_f16<<<NIN / (256 * 4), 256, 0, stream>>>(value, vH, NIN);
  cvt_f32_to_f16<<<NW  / (256 * 4), 256, 0, stream>>>(Wq, WqH, NW);
  cvt_f32_to_f16<<<NW  / (256 * 4), 256, 0, stream>>>(Wk, WkH, NW);
  cvt_f32_to_f16<<<NW  / (256 * 4), 256, 0, stream>>>(Wv, WvH, NW);
  cvt_f32_to_f16<<<NW  / (256 * 4), 256, 0, stream>>>(Wo, WoH, NW);

  // 2) QKV projections (Dh^-0.5 folded into Q as in the reference)
  const int projBlocks = (M / 16) * (E_DIM / 64) / 8;      // 512
  proj_gemm<<<projBlocks, 256, 0, stream>>>(qH, WqH, bq, 0.125f, Qp, nullptr, M);
  proj_gemm<<<projBlocks, 256, 0, stream>>>(kH, WkH, bk, 1.0f,   Kp, nullptr, M);
  proj_gemm<<<projBlocks, 256, 0, stream>>>(vH, WvH, bv, 1.0f,   Vp, nullptr, M);

  // 3) attention: QK^T -> softmax (weights streamed NT to d_out) -> P@V
  const size_t ldsBytes = 16 * S_LEN * 4 + 16 * S_LEN * 2 + 16 * 64 * 4
                        + 2 * VCHUNK * 64 * 2;             // 266240 B <= 320K
  attn_kernel<<<B_NUM * H_NUM * (S_LEN / 16), 256, ldsBytes, stream>>>(
      Qp, Kp, Vp, out_w, Ctx);

  // 4) output projection to f32
  proj_gemm<<<projBlocks, 256, 0, stream>>>(Ctx, WoH, bo, 1.0f, nullptr, out_attn, M);
}
